// DSAGPredictor_11312943857890
// MI455X (gfx1250) — matrix operand order repaired and checked by reading, not compiled, not run on hardware
//
#include <hip/hip_runtime.h>
#include <hip/hip_bf16.h>

#define NTOK 9216      // 96*96 tokens
#define CDIM 256       // channels
#define ZREP 16
#define L2E  1.44269504088896340736f

typedef __attribute__((ext_vector_type(16))) __bf16 v16bf;
typedef __attribute__((ext_vector_type(8)))  float  v8f;

union FragBF { v16bf v; uint4 q[2]; };

// ---- f32 -> bf16 (round to nearest even) ----
__device__ __forceinline__ unsigned short bf16b(float f) {
    unsigned u = __builtin_bit_cast(unsigned, f);
    u += 0x7fffu + ((u >> 16) & 1u);
    return (unsigned short)(u >> 16);
}
__device__ __forceinline__ unsigned int pk2(float a, float b) {
    return (unsigned int)bf16b(a) | ((unsigned int)bf16b(b) << 16);
}

// A fragment (16x32, row-major [m][k] bf16): lane m=lane&15, halves 0..7 = K hi*8+.,
// halves 8..15 = K 16+hi*8+.  (ISA 7.12.2, 16-bit A 16x32)
__device__ __forceinline__ v16bf ldfragA(const unsigned short* p, int stride, int lane) {
    int m = lane & 15, hi = lane >> 4;
    const unsigned short* r = p + m * stride + hi * 8;
    FragBF f;
    f.q[0] = *(const uint4*)(r);
    f.q[1] = *(const uint4*)(r + 16);
    return f.v;
}
// B fragment (32x16 K x N), stored B-transposed [n][k] bf16: lane n=lane&15 holds
// halves 0..7 = K hi*8+., halves 8..15 = K 16+hi*8+. (matches A fragment order)
__device__ __forceinline__ v16bf ldfragB(const unsigned short* p, int stride, int lane) {
    int n = lane & 15, hi = lane >> 4;
    const unsigned short* r = p + n * stride + hi * 8;
    FragBF f;
    f.q[0] = *(const uint4*)(r);
    f.q[1] = *(const uint4*)(r + 16);
    return f.v;
}
__device__ __forceinline__ v8f wmma_bf16(v16bf a, v16bf b, v8f c) {
    return __builtin_amdgcn_wmma_f32_16x16x32_bf16(false, a, false, b, (short)0, c, false, false);
}
__device__ __forceinline__ v8f vzero8() {
    v8f v;
#pragma unroll
    for (int i = 0; i < 8; ++i) v[i] = 0.0f;
    return v;
}

// ============================================================================
// Kernel 1: q/k/v = W{q,k,v} @ xf.  blockIdx.y selects weight.
// Outputs: Qt,Kt bf16 token-major [n][c]; Vb bf16 channel-major [c][n].
// ============================================================================
__global__ __launch_bounds__(256) void qkv_kernel(
    const float* __restrict__ x,
    const float* __restrict__ Wq, const float* __restrict__ Wk, const float* __restrict__ Wv,
    unsigned short* __restrict__ Qt, unsigned short* __restrict__ Kt,
    unsigned short* __restrict__ Vb)
{
    const int wi = blockIdx.y;
    const float* W = (wi == 0) ? Wq : (wi == 1) ? Wk : Wv;
    const int tb = blockIdx.x * 128;
    const int t = threadIdx.x, wave = t >> 5, lane = t & 31;
    const int hi = lane >> 4, n0 = lane & 15;

    __shared__ unsigned short sA[256][40];   // W chunk  [o][kc]  (bf16, padded)
    __shared__ unsigned short sB[128][40];   // x chunk  [n][kc]  (transposed, bf16)

    v8f acc[16];
#pragma unroll
    for (int i = 0; i < 16; ++i) acc[i] = vzero8();

    for (int kc = 0; kc < CDIM; kc += 32) {
        __syncthreads();
        // stage A: 256x32 weight slice (rows contiguous)
#pragma unroll
        for (int i = 0; i < 32; ++i) {
            int idx = t + i * 256, row = idx >> 5, col = idx & 31;
            sA[row][col] = bf16b(W[row * CDIM + kc + col]);
        }
        // stage B: 32x128 x-slice, transposed to [n][c]
#pragma unroll
        for (int i = 0; i < 16; ++i) {
            int idx = t + i * 256, c = idx >> 7, n = idx & 127;
            sB[n][c] = bf16b(x[(kc + c) * NTOK + tb + n]);
        }
        __syncthreads();

        v16bf a0 = ldfragA(&sA[(wave * 2 + 0) * 16][0], 40, lane);
        v16bf a1 = ldfragA(&sA[(wave * 2 + 1) * 16][0], 40, lane);
        // depth-2 pipelined B fragments
        v16bf bcur = ldfragB(&sB[0][0], 40, lane);
#pragma unroll
        for (int nt = 0; nt < 8; ++nt) {
            v16bf bnxt;
            if (nt < 7) bnxt = ldfragB(&sB[(nt + 1) * 16][0], 40, lane);
            acc[0 * 8 + nt] = wmma_bf16(a0, bcur, acc[0 * 8 + nt]);
            acc[1 * 8 + nt] = wmma_bf16(a1, bcur, acc[1 * 8 + nt]);
            bcur = bnxt;
        }
    }
    // epilogue
#pragma unroll
    for (int mt = 0; mt < 2; ++mt) {
#pragma unroll
        for (int nt = 0; nt < 8; ++nt) {
            v8f v = acc[mt * 8 + nt];
            int c0 = (wave * 2 + mt) * 16 + hi * 8;
            int n  = tb + nt * 16 + n0;
            if (wi < 2) {
                unsigned short* dst = ((wi == 0) ? Qt : Kt) + n * CDIM + c0;
                uint4 o;
                o.x = pk2(v[0], v[1]); o.y = pk2(v[2], v[3]);
                o.z = pk2(v[4], v[5]); o.w = pk2(v[6], v[7]);
                *(uint4*)dst = o;
            } else {
#pragma unroll
                for (int r = 0; r < 8; ++r)
                    Vb[(c0 + r) * NTOK + n] = bf16b(v[r]);
            }
        }
    }
}

// ============================================================================
// Kernel 2: rel = embd[isWithin, dist+8];  pos[z][n] = sum_c rel[z][c]*x[c][n]
// ============================================================================
__global__ __launch_bounds__(256) void pos_kernel(
    const float* __restrict__ x, const float* __restrict__ embd,
    const int* __restrict__ dist, const int* __restrict__ isW,
    float* __restrict__ pos)
{
    __shared__ float rel[ZREP][257];
    int t = threadIdx.x;
    for (int i = t; i < ZREP * CDIM; i += 256) {
        int z = i >> 8, c = i & 255;
        int off = isW[z] * (17 * CDIM) + (dist[z] + 8) * CDIM + c;
        rel[z][c] = embd[off];
    }
    __syncthreads();
    int n = blockIdx.x * 256 + t;
    float acc[ZREP];
#pragma unroll
    for (int z = 0; z < ZREP; ++z) acc[z] = 0.0f;
    for (int c = 0; c < CDIM; ++c) {
        float xv = x[c * NTOK + n];
#pragma unroll
        for (int z = 0; z < ZREP; ++z) acc[z] = fmaf(rel[z][c], xv, acc[z]);
    }
#pragma unroll
    for (int z = 0; z < ZREP; ++z) pos[z * NTOK + n] = acc[z];
}

// ============================================================================
// Kernel 3: pass 1 — per-k-row online (max, sum-exp) over all q of s = K^T Q.
// Writes l2r[k] = max*log2e + log2(sum) so that p = exp2(s*log2e - l2r).
// ============================================================================
__global__ __launch_bounds__(256) void rownorm_kernel(
    const unsigned short* __restrict__ Qt, const unsigned short* __restrict__ Kt,
    float* __restrict__ l2r)
{
    const int t = threadIdx.x, wave = t >> 5, lane = t & 31;
    const int hi = lane >> 4, n0 = lane & 15;
    const int kb = blockIdx.x * 128;

    // preload this wave's K^T A-fragments (16 k-rows x full C)
    v16bf afr[8];
#pragma unroll
    for (int cs = 0; cs < 8; ++cs)
        afr[cs] = ldfragA(Kt + (kb + wave * 16) * CDIM + cs * 32, CDIM, lane);

    float mmax[8], rs[8];
#pragma unroll
    for (int r = 0; r < 8; ++r) { mmax[r] = -3.0e38f; rs[r] = 0.0f; }

    for (int q0 = 0; q0 < NTOK; q0 += 16) {
        // prefetch next q-tile rows into WGP$/L2 (global_prefetch_b8)
        __builtin_prefetch(Qt + (q0 + 16) * CDIM + lane * 16, 0, 0);
        // preload all 8 B fragments, then chain WMMAs (loads overlap matrix pipe)
        v16bf bfr[8];
#pragma unroll
        for (int cs = 0; cs < 8; ++cs)
            bfr[cs] = ldfragB(Qt + q0 * CDIM + cs * 32, CDIM, lane);
        v8f s = vzero8();
#pragma unroll
        for (int cs = 0; cs < 8; ++cs)
            s = wmma_bf16(afr[cs], bfr[cs], s);
#pragma unroll
        for (int r = 0; r < 8; ++r) {
            float sv = s[r];
            float mn = fmaxf(mmax[r], sv);
            rs[r] = rs[r] * exp2f((mmax[r] - mn) * L2E) + exp2f((sv - mn) * L2E);
            mmax[r] = mn;
        }
    }
    // cross-lane reduce: 16 lanes share each row
    __shared__ float smx[128][17];
    __shared__ float ssm[128][17];
#pragma unroll
    for (int r = 0; r < 8; ++r) {
        int row = wave * 16 + hi * 8 + r;
        smx[row][n0] = mmax[r];
        ssm[row][n0] = rs[r];
    }
    __syncthreads();
    if (t < 128) {
        float m = -3.0e38f;
#pragma unroll
        for (int i = 0; i < 16; ++i) m = fmaxf(m, smx[t][i]);
        float s = 0.0f;
#pragma unroll
        for (int i = 0; i < 16; ++i) s += ssm[t][i] * exp2f((smx[t][i] - m) * L2E);
        l2r[kb + t] = m * L2E + log2f(s);
    }
}

// ============================================================================
// Kernel 4: pass 2 — flash-style: recompute S tiles, P = exp2(S*log2e - l2r)
// into LDS (bf16, [q][k] layout), accumulate Y = Vb @ P, add residual x,
// write out_t bf16 token-major [n][c].
// ============================================================================
__global__ __launch_bounds__(256) void attn_out_kernel(
    const unsigned short* __restrict__ Qt, const unsigned short* __restrict__ Kt,
    const unsigned short* __restrict__ Vb, const float* __restrict__ l2r,
    const float* __restrict__ x, unsigned short* __restrict__ out_t)
{
    const int t = threadIdx.x, wave = t >> 5, lane = t & 31;
    const int hi = lane >> 4, n0 = lane & 15;
    const int qb = blockIdx.x * 128;

    __shared__ unsigned short sP[128][40];   // P chunk [q][k] bf16
    __shared__ float sl2r[32];

    // wave w's q-subtile B fragments (fixed for whole k loop)
    v16bf qfr[8];
#pragma unroll
    for (int cs = 0; cs < 8; ++cs)
        qfr[cs] = ldfragB(Qt + (qb + wave * 16) * CDIM + cs * 32, CDIM, lane);

    v8f acc[16];
#pragma unroll
    for (int i = 0; i < 16; ++i) acc[i] = vzero8();

    for (int kc = 0; kc < NTOK; kc += 32) {
        if (t < 32) sl2r[t] = l2r[kc + t];
        // prefetch next chunk of K^T and V rows (global_prefetch_b8)
        __builtin_prefetch(Kt + (kc + 32 + lane) * CDIM, 0, 0);
        __builtin_prefetch(Vb + (wave * 32 + lane) * NTOK + kc + 32, 0, 0);
        __syncthreads();
        // S phase: wave w computes [32k x 16q] for its q-subtile, then P -> LDS
#pragma unroll
        for (int mt = 0; mt < 2; ++mt) {
            v8f s = vzero8();
            // depth-2 pipelined A fragments from global Kt
            v16bf a_cur = ldfragA(Kt + (kc + mt * 16) * CDIM, CDIM, lane);
#pragma unroll
            for (int cs = 0; cs < 8; ++cs) {
                v16bf a_nxt;
                if (cs < 7) a_nxt = ldfragA(Kt + (kc + mt * 16) * CDIM + (cs + 1) * 32, CDIM, lane);
                s = wmma_bf16(a_cur, qfr[cs], s);
                a_cur = a_nxt;
            }
            float p[8];
#pragma unroll
            for (int r = 0; r < 8; ++r)
                p[r] = exp2f(s[r] * L2E - sl2r[mt * 16 + hi * 8 + r]);
            uint4 o;
            o.x = pk2(p[0], p[1]); o.y = pk2(p[2], p[3]);
            o.z = pk2(p[4], p[5]); o.w = pk2(p[6], p[7]);
            *(uint4*)&sP[wave * 16 + n0][mt * 16 + hi * 8] = o;
        }
        __syncthreads();
        // Y phase: wave w owns c-rows [32w, 32w+32) across all 8 q-subtiles
        v16bf a0 = ldfragA(Vb + (wave * 32 + 0)  * NTOK + kc, NTOK, lane);
        v16bf a1 = ldfragA(Vb + (wave * 32 + 16) * NTOK + kc, NTOK, lane);
        // depth-2 pipelined B fragments from LDS sP
        v16bf bcur = ldfragB(&sP[0][0], 40, lane);
#pragma unroll
        for (int nt = 0; nt < 8; ++nt) {
            v16bf bnxt;
            if (nt < 7) bnxt = ldfragB(&sP[(nt + 1) * 16][0], 40, lane);
            acc[0 * 8 + nt] = wmma_bf16(a0, bcur, acc[0 * 8 + nt]);
            acc[1 * 8 + nt] = wmma_bf16(a1, bcur, acc[1 * 8 + nt]);
            bcur = bnxt;
        }
    }
    // epilogue: residual add, store out_t [n][c] bf16
#pragma unroll
    for (int mt = 0; mt < 2; ++mt) {
#pragma unroll
        for (int nt = 0; nt < 8; ++nt) {
            v8f v = acc[mt * 8 + nt];
            int c0 = wave * 32 + mt * 16 + hi * 8;
            int n  = qb + nt * 16 + n0;
            float o[8];
#pragma unroll
            for (int r = 0; r < 8; ++r) o[r] = v[r] + x[(c0 + r) * NTOK + n];
            uint4 w4;
            w4.x = pk2(o[0], o[1]); w4.y = pk2(o[2], o[3]);
            w4.z = pk2(o[4], o[5]); w4.w = pk2(o[6], o[7]);
            *(uint4*)(out_t + n * CDIM + c0) = w4;
        }
    }
}

// ============================================================================
// Kernel 5: base = Wproj[:, :256] @ out;  result[z,o,n] = base + Wproj[o,256]*pos[z,n]
// Writes the full [16,256,9216] f32 output.
// ============================================================================
__global__ __launch_bounds__(256) void proj_kernel(
    const float* __restrict__ Wproj, const unsigned short* __restrict__ out_t,
    const float* __restrict__ pos, float* __restrict__ outp)
{
    const int t = threadIdx.x, wave = t >> 5, lane = t & 31;
    const int hi = lane >> 4, n0 = lane & 15;
    const int tb = blockIdx.x * 128;

    __shared__ unsigned short sA[256][40];
    __shared__ float wlast[256];
    wlast[t] = Wproj[t * (CDIM + 1) + CDIM];

    v8f acc[16];
#pragma unroll
    for (int i = 0; i < 16; ++i) acc[i] = vzero8();

    for (int kc = 0; kc < CDIM; kc += 32) {
        __syncthreads();
#pragma unroll
        for (int i = 0; i < 32; ++i) {
            int idx = t + i * 256, row = idx >> 5, col = idx & 31;
            sA[row][col] = bf16b(Wproj[row * (CDIM + 1) + kc + col]);
        }
        __syncthreads();
        v16bf a0 = ldfragA(&sA[(wave * 2 + 0) * 16][0], 40, lane);
        v16bf a1 = ldfragA(&sA[(wave * 2 + 1) * 16][0], 40, lane);
        v16bf bcur = ldfragB(out_t + (tb + 0) * CDIM + kc, CDIM, lane);
#pragma unroll
        for (int nt = 0; nt < 8; ++nt) {
            v16bf bnxt;
            if (nt < 7) bnxt = ldfragB(out_t + (tb + (nt + 1) * 16) * CDIM + kc, CDIM, lane);
            acc[0 * 8 + nt] = wmma_bf16(a0, bcur, acc[0 * 8 + nt]);
            acc[1 * 8 + nt] = wmma_bf16(a1, bcur, acc[1 * 8 + nt]);
            bcur = bnxt;
        }
    }
    // fused z-broadcast epilogue
#pragma unroll
    for (int mt = 0; mt < 2; ++mt) {
#pragma unroll
        for (int nt = 0; nt < 8; ++nt) {
            v8f v = acc[mt * 8 + nt];
            int o0 = (wave * 2 + mt) * 16 + hi * 8;
            int n  = tb + nt * 16 + n0;
            float wl[8];
#pragma unroll
            for (int r = 0; r < 8; ++r) wl[r] = wlast[o0 + r];
#pragma unroll
            for (int z = 0; z < ZREP; ++z) {
                float pz = pos[z * NTOK + n];
                float* dst = outp + (size_t)z * CDIM * NTOK + n;
#pragma unroll
                for (int r = 0; r < 8; ++r)
                    dst[(size_t)(o0 + r) * NTOK] = fmaf(wl[r], pz, v[r]);
            }
        }
    }
}

// ============================================================================
extern "C" void kernel_launch(void* const* d_in, const int* in_sizes, int n_in,
                              void* d_out, int out_size, void* d_ws, size_t ws_size,
                              hipStream_t stream) {
    const float* x     = (const float*)d_in[0];
    const float* Wq    = (const float*)d_in[1];
    const float* Wk    = (const float*)d_in[2];
    const float* Wv    = (const float*)d_in[3];
    const float* embd  = (const float*)d_in[4];
    const float* Wproj = (const float*)d_in[5];
    const int*   dist  = (const int*)d_in[6];
    const int*   isW   = (const int*)d_in[7];
    float* outp = (float*)d_out;

    char* ws = (char*)d_ws;
    size_t off = 0;
    unsigned short* Qt    = (unsigned short*)(ws + off); off += (size_t)NTOK * CDIM * 2;  // 4.72 MB
    unsigned short* Kt    = (unsigned short*)(ws + off); off += (size_t)NTOK * CDIM * 2;
    unsigned short* Vb    = (unsigned short*)(ws + off); off += (size_t)NTOK * CDIM * 2;
    unsigned short* out_t = (unsigned short*)(ws + off); off += (size_t)NTOK * CDIM * 2;
    float* l2r            = (float*)(ws + off);          off += (size_t)NTOK * 4;
    float* pos            = (float*)(ws + off);          off += (size_t)ZREP * NTOK * 4;

    qkv_kernel<<<dim3(NTOK / 128, 3), 256, 0, stream>>>(x, Wq, Wk, Wv, Qt, Kt, Vb);
    pos_kernel<<<NTOK / 256, 256, 0, stream>>>(x, embd, dist, isW, pos);
    rownorm_kernel<<<NTOK / 128, 256, 0, stream>>>(Qt, Kt, l2r);
    attn_out_kernel<<<NTOK / 128, 256, 0, stream>>>(Qt, Kt, Vb, l2r, x, out_t);
    proj_kernel<<<NTOK / 128, 256, 0, stream>>>(Wproj, out_t, pos, outp);
}
